// Rot_MultiheadAttention_70798240907936
// MI455X (gfx1250) — matrix operand order repaired and checked by reading, not compiled, not run on hardware
//
#include <hip/hip_runtime.h>
#include <cstdint>
#include <cstddef>

typedef __attribute__((ext_vector_type(16))) __bf16 v16bf;
typedef __attribute__((ext_vector_type(8)))  float  v8f;

#define HID    512
#define HEADS  8
#define HD     64
#define S_LEN  4096
#define BATCH  2
// ln(1e8)
#define LOG_THETA 18.420680743952367f

union FragB16 { v16bf v; unsigned int u[8]; };

// Async copy: 16 bytes global -> LDS, tracked by ASYNCcnt (CDNA5).
__device__ __forceinline__ void async_copy_b128(uint32_t lds_off, const void* gaddr)
{
  asm volatile("global_load_async_to_lds_b128 %0, %1, off"
               :: "v"(lds_off), "v"((uint64_t)(uintptr_t)gaddr)
               : "memory");
}
__device__ __forceinline__ void wait_asynccnt0()
{
  asm volatile("s_wait_asynccnt 0x0" ::: "memory");
}

// ---------------------------------------------------------------------------
// Kernel 1: projection GEMM (out = X @ W^T + b), fused RoPE, pack to bf16.
// grid (128, 8, 3); block = 128 (4 waves); each wave owns a 16x64 strip.
// ---------------------------------------------------------------------------
__global__ void proj_rope_kernel(const float* __restrict__ q_in,
                                 const float* __restrict__ k_in,
                                 const float* __restrict__ v_in,
                                 const float* __restrict__ Wq, const float* __restrict__ bq,
                                 const float* __restrict__ Wk, const float* __restrict__ bk,
                                 const float* __restrict__ Wv, const float* __restrict__ bv,
                                 __bf16* __restrict__ qbf,   // [b][h][s][d]
                                 __bf16* __restrict__ kbf,   // [b][h][s][d]
                                 __bf16* __restrict__ vt)    // [b][h][d][s]
{
  const int z = blockIdx.z;
  const float* X    = (z == 0) ? q_in : (z == 1) ? k_in : v_in;
  const float* W    = (z == 0) ? Wq   : (z == 1) ? Wk   : Wv;
  const float* bias = (z == 0) ? bq   : (z == 1) ? bk   : bv;

  const int mtile = blockIdx.x;   // 0..127
  const int ntile = blockIdx.y;   // 0..7
  const int tid   = threadIdx.x;
  const int wave  = tid >> 5;
  const int lane  = tid & 31;
  const int lhalf = lane & 15;
  const int hi    = lane >> 4;

  __shared__ alignas(16) __bf16 At[64 * 34];   // [m][k], stride 34
  __shared__ alignas(16) __bf16 Bt[64 * 34];   // [n][k] = W[jbase+n][k0+k]

  v8f acc[4] = {};

  const int rowbase = mtile * 64;
  const int jbase   = ntile * 64;

  for (int k0 = 0; k0 < HID; k0 += 32) {
    {
      const int r  = tid >> 1;
      const int kh = (tid & 1) * 16;
      const float* src  = X + (size_t)(rowbase + r) * HID + k0 + kh;
      const float* wsrc = W + (size_t)(jbase   + r) * HID + k0 + kh;
#pragma unroll
      for (int i = 0; i < 16; ++i) At[r * 34 + kh + i] = (__bf16)src[i];
#pragma unroll
      for (int i = 0; i < 16; ++i) Bt[r * 34 + kh + i] = (__bf16)wsrc[i];
    }
    __syncthreads();

    // A fragment 16x32 (lanes 0-15 hold K 0-7 & 16-23)
    FragB16 a;
    const int m = wave * 16 + lhalf;
#pragma unroll
    for (int v = 0; v < 8; ++v) {
      const int kk = ((v >= 4) ? 16 : 0) + hi * 8 + (v & 3) * 2;
      a.u[v] = *(const unsigned int*)&At[m * 34 + kk];
    }
#pragma unroll
    for (int ns = 0; ns < 4; ++ns) {
      // B fragment 32x16 (lanes 0-15 hold K 0-7 & 16-23)
      FragB16 bfr;
      const int n = ns * 16 + lhalf;
#pragma unroll
      for (int v = 0; v < 8; ++v) {
        const int kk = ((v >= 4) ? 16 : 0) + hi * 8 + (v & 3) * 2;
        bfr.u[v] = *(const unsigned int*)&Bt[n * 34 + kk];
      }
      acc[ns] = __builtin_amdgcn_wmma_f32_16x16x32_bf16(
          false, a.v, false, bfr.v, (short)0, acc[ns], false, false);
    }
    __syncthreads();
  }

  // Epilogue: bias + (RoPE for z<2) + pack bf16, scatter to scratch.
  const int gm0 = rowbase + wave * 16;
#pragma unroll
  for (int ns = 0; ns < 4; ++ns) {
#pragma unroll
    for (int v = 0; v < 8; ++v) {
      const int mrow = v + hi * 8;
      const int grow = gm0 + mrow;
      const int b    = grow >> 12;
      const int s    = grow & 4095;
      const int j    = jbase + ns * 16 + lhalf;
      float val = acc[ns][v] + bias[j];
      if (z < 2) {
        const float other = __shfl_xor(val, 1);
        const int p    = j >> 1;
        const int fi   = p >> 1;
        const int axis = p & 1;
        const float pos = (float)(axis ? (s & 63) : (s >> 6));
        const float ang = pos * __expf(-(float)fi * (LOG_THETA / 128.0f));
        float sn, cs;
        __sincosf(ang, &sn, &cs);
        val = (j & 1) ? (other * sn + val * cs)
                      : (val * cs - other * sn);
      }
      const int hh = j >> 6;
      const int d  = j & 63;
      if (z == 0)
        qbf[((size_t)(b * HEADS + hh) * S_LEN + s) * HD + d] = (__bf16)val;
      else if (z == 1)
        kbf[((size_t)(b * HEADS + hh) * S_LEN + s) * HD + d] = (__bf16)val;
      else
        vt[((size_t)(b * HEADS + hh) * HD + d) * S_LEN + s] = (__bf16)val;
    }
  }
}

// ---------------------------------------------------------------------------
// Kernel 2: flash attention per (b,h). grid (64, 16), block 128 (4 waves).
// 64-key blocks staged into LDS via GLOBAL_LOAD_ASYNC_TO_LDS_B128 (ASYNCcnt),
// then all 4 waves run 16 WMMAs per block from conflict-free LDS tiles.
// ---------------------------------------------------------------------------
#define KROW 72   // K/V LDS row stride in bf16 elems (144 B -> 36 dw, no bank conflicts)
#define PROW 66   // P  LDS row stride in bf16 elems (132 B -> 33 dw)

__global__ void flash_attn_kernel(const __bf16* __restrict__ qbf,
                                  const __bf16* __restrict__ kbf,
                                  const __bf16* __restrict__ vt,
                                  float* __restrict__ out)
{
  const int qtile = blockIdx.x;   // 0..63
  const int bh    = blockIdx.y;   // 0..15
  const int b     = bh >> 3;
  const int h     = bh & 7;
  const int tid   = threadIdx.x;
  const int wave  = tid >> 5;
  const int lane  = tid & 31;
  const int lhalf = lane & 15;
  const int hi    = lane >> 4;

  __shared__ alignas(16) __bf16 Kt[64 * KROW];    // [key][d]
  __shared__ alignas(16) __bf16 Vl[64 * KROW];    // [d][key]
  __shared__ alignas(16) __bf16 Pt[4][16 * PROW]; // per-wave P staging

  const uint32_t* q32 = (const uint32_t*)qbf;

  const int    srow0  = qtile * 64 + wave * 16;
  const size_t qkbase = (size_t)bh * S_LEN * HD;   // bf16-element offset
  const size_t vbase  = (size_t)bh * HD * S_LEN;

  // Q as two 16x32 A-fragments (d 0..31 and 32..63), loaded once.
  FragB16 qa0, qa1;
  {
    const int s = srow0 + lhalf;
#pragma unroll
    for (int v = 0; v < 8; ++v) {
      const int d = ((v >= 4) ? 16 : 0) + hi * 8 + (v & 3) * 2;
      qa0.u[v] = q32[(qkbase + (size_t)s * HD + d) >> 1];
      qa1.u[v] = q32[(qkbase + (size_t)s * HD + d + 32) >> 1];
    }
  }

  v8f  o[4] = {};
  float rm[8], rl[8];
#pragma unroll
  for (int v = 0; v < 8; ++v) { rm[v] = -INFINITY; rl[v] = 0.0f; }

  const float scale = 1.0f / 64.0f;   // reference divides by head_dim

  for (int kb = 0; kb < S_LEN / 64; ++kb) {
    const int key0 = kb * 64;
    if (kb + 1 < S_LEN / 64)           // warm L2 for next block
      __builtin_prefetch(&kbf[qkbase + (size_t)(key0 + 64) * HD], 0, 2);

    // ---- async-stage K (64x64) and V (64x64) blocks into LDS, once per WG
#pragma unroll
    for (int t = 0; t < 4; ++t) {
      const int chunk = tid + t * 128;   // 0..511
      const int row   = chunk >> 3;      // 0..63
      const int seg   = chunk & 7;       // 8 elems (16 B) each
      async_copy_b128((uint32_t)(uintptr_t)&Kt[row * KROW + seg * 8],
                      kbf + qkbase + (size_t)(key0 + row) * HD + seg * 8);
      async_copy_b128((uint32_t)(uintptr_t)&Vl[row * KROW + seg * 8],
                      vt + vbase + (size_t)row * S_LEN + key0 + seg * 8);
    }
    wait_asynccnt0();
    __syncthreads();

    // ---- S = Q K^T : four 16x16 C tiles (K-dim 64 = 2 WMMAs each)
    v8f sv[4];
#pragma unroll
    for (int ns = 0; ns < 4; ++ns) {
      FragB16 b0, b1;
      const int keyl = ns * 16 + lhalf;
#pragma unroll
      for (int v = 0; v < 8; ++v) {
        const int d = ((v >= 4) ? 16 : 0) + hi * 8 + (v & 3) * 2;
        b0.u[v] = *(const unsigned int*)&Kt[keyl * KROW + d];
        b1.u[v] = *(const unsigned int*)&Kt[keyl * KROW + d + 32];
      }
      v8f acc = {};
      acc = __builtin_amdgcn_wmma_f32_16x16x32_bf16(false, qa0.v, false, b0.v,
                                                    (short)0, acc, false, false);
      acc = __builtin_amdgcn_wmma_f32_16x16x32_bf16(false, qa1.v, false, b1.v,
                                                    (short)0, acc, false, false);
      sv[ns] = acc;
    }

    // ---- online softmax over 64 keys (row stats per 16-lane half)
    float p[4][8], alpha[8];
#pragma unroll
    for (int v = 0; v < 8; ++v) {
      float a0 = sv[0][v] * scale, a1 = sv[1][v] * scale;
      float a2 = sv[2][v] * scale, a3 = sv[3][v] * scale;
      float t = fmaxf(fmaxf(a0, a1), fmaxf(a2, a3));
      t = fmaxf(t, __shfl_xor(t, 1));
      t = fmaxf(t, __shfl_xor(t, 2));
      t = fmaxf(t, __shfl_xor(t, 4));
      t = fmaxf(t, __shfl_xor(t, 8));
      const float mn = fmaxf(rm[v], t);
      p[0][v] = __expf(a0 - mn);
      p[1][v] = __expf(a1 - mn);
      p[2][v] = __expf(a2 - mn);
      p[3][v] = __expf(a3 - mn);
      float rs = (p[0][v] + p[1][v]) + (p[2][v] + p[3][v]);
      rs += __shfl_xor(rs, 1);
      rs += __shfl_xor(rs, 2);
      rs += __shfl_xor(rs, 4);
      rs += __shfl_xor(rs, 8);
      alpha[v] = __expf(rm[v] - mn);
      rl[v] = rl[v] * alpha[v] + rs;
      rm[v] = mn;
    }
#pragma unroll
    for (int ds = 0; ds < 4; ++ds)
#pragma unroll
      for (int v = 0; v < 8; ++v) o[ds][v] *= alpha[v];

    // ---- C-layout P(16x64) -> two A-fragments via per-wave LDS tile
    __bf16* P = Pt[wave];
#pragma unroll
    for (int ns = 0; ns < 4; ++ns)
#pragma unroll
      for (int v = 0; v < 8; ++v)
        P[(v + hi * 8) * PROW + ns * 16 + lhalf] = (__bf16)p[ns][v];

    FragB16 pa0, pa1;
#pragma unroll
    for (int v = 0; v < 8; ++v) {
      const int kk = ((v >= 4) ? 16 : 0) + hi * 8 + (v & 3) * 2;
      pa0.u[v] = *(const unsigned int*)&P[lhalf * PROW + kk];
      pa1.u[v] = *(const unsigned int*)&P[lhalf * PROW + kk + 32];
    }

    // ---- O += P * V  (V tile is [d][key] in LDS: key pairs contiguous)
#pragma unroll
    for (int ds = 0; ds < 4; ++ds) {
      FragB16 vb0, vb1;
      const int d = ds * 16 + lhalf;
#pragma unroll
      for (int v = 0; v < 8; ++v) {
        const int keyl = ((v >= 4) ? 16 : 0) + hi * 8 + (v & 3) * 2;
        vb0.u[v] = *(const unsigned int*)&Vl[d * KROW + keyl];
        vb1.u[v] = *(const unsigned int*)&Vl[d * KROW + keyl + 32];
      }
      o[ds] = __builtin_amdgcn_wmma_f32_16x16x32_bf16(false, pa0.v, false, vb0.v,
                                                      (short)0, o[ds], false, false);
      o[ds] = __builtin_amdgcn_wmma_f32_16x16x32_bf16(false, pa1.v, false, vb1.v,
                                                      (short)0, o[ds], false, false);
    }

    __syncthreads();   // all waves done reading Kt/Vl before next async stage
  }

  // ---- normalize and store: out[b][s][h*64+d]
#pragma unroll
  for (int ds = 0; ds < 4; ++ds) {
#pragma unroll
    for (int v = 0; v < 8; ++v) {
      const int mrow = v + hi * 8;
      const int s = srow0 + mrow;
      const int d = ds * 16 + lhalf;
      out[((size_t)(b * S_LEN + s)) * HID + h * HD + d] = o[ds][v] / rl[v];
    }
  }
}

// ---------------------------------------------------------------------------
extern "C" void kernel_launch(void* const* d_in, const int* in_sizes, int n_in,
                              void* d_out, int out_size, void* d_ws, size_t ws_size,
                              hipStream_t stream)
{
  (void)in_sizes; (void)n_in; (void)out_size; (void)ws_size;
  const float* q_in = (const float*)d_in[0];
  const float* k_in = (const float*)d_in[1];
  const float* v_in = (const float*)d_in[2];
  const float* Wq   = (const float*)d_in[3];
  const float* bq   = (const float*)d_in[4];
  const float* Wk   = (const float*)d_in[5];
  const float* bk   = (const float*)d_in[6];
  const float* Wv   = (const float*)d_in[7];
  const float* bv   = (const float*)d_in[8];

  const size_t elems = (size_t)BATCH * HEADS * S_LEN * HD;   // 4 Mi elements
  __bf16* qbf = (__bf16*)d_ws;
  __bf16* kbf = qbf + elems;
  __bf16* vt  = kbf + elems;   // 24 MB total bf16 scratch

  dim3 gA(128, 8, 3), bA(128);
  proj_rope_kernel<<<gA, bA, 0, stream>>>(q_in, k_in, v_in, Wq, bq, Wk, bk,
                                          Wv, bv, qbf, kbf, vt);

  dim3 gB(64, 16), bB(128);
  flash_attn_kernel<<<gB, bB, 0, stream>>>(qbf, kbf, vt, (float*)d_out);
}
